// SpectralWeighting_44813688767179
// MI455X (gfx1250) — compile-verified, matrix operand order checked by reading, not run on hardware
//
#include <hip/hip_runtime.h>
#include <hip/hip_bf16.h>

// ---------------------------------------------------------------------------
// ChebConv (K=3) + LayerNorm, MI455X (gfx1250, wave32, WMMA bf16)
// Only layer L-1 is live (reference overwrites x each layer with cheb(feats)).
// Props are CSR-gather (no float atomics); GEMM is 64-row WMMA blocks with
// B-fragment register reuse.
// ---------------------------------------------------------------------------

typedef __attribute__((ext_vector_type(16))) __bf16 v16bf;
typedef __attribute__((ext_vector_type(8)))  __bf16 v8bf;
typedef __attribute__((ext_vector_type(8)))  float  v8f;

#define NB    8
#define NN    20000
#define CD    256
#define NE    320000
#define KORD  3
#define NROWS (NB * NN)          /* 160000 */
#define EPSF  1e-5f

// ------------------------------- utility kernels ---------------------------

__global__ __launch_bounds__(256) void k_zero_f4(float4* p, long n4) {
    long i = (long)blockIdx.x * blockDim.x + threadIdx.x;
    if (i < n4) p[i] = make_float4(0.f, 0.f, 0.f, 0.f);
}

__global__ __launch_bounds__(256) void k_zero_i(int* p, int n) {
    int i = blockIdx.x * blockDim.x + threadIdx.x;
    if (i < n) p[i] = 0;
}

__global__ __launch_bounds__(256) void k_degree(const long long* __restrict__ ei,
                                                float* __restrict__ deg) {
    int e = blockIdx.x * blockDim.x + threadIdx.x;
    if (e < NE) atomicAdd(&deg[(int)ei[e]], 1.0f);
}

__global__ __launch_bounds__(256) void k_norm(const long long* __restrict__ ei,
                                              const float* __restrict__ deg,
                                              float* __restrict__ nrm) {
    int e = blockIdx.x * blockDim.x + threadIdx.x;
    if (e >= NE) return;
    int s = (int)ei[e];
    int d = (int)ei[NE + e];
    float ds = deg[s], dd = deg[d];
    float a = ds > 0.f ? rsqrtf(ds) : 0.f;
    float c = dd > 0.f ? rsqrtf(dd) : 0.f;
    nrm[e] = -a * c;
}

// ------------------------------- CSR build (by dst) ------------------------

__global__ __launch_bounds__(256) void k_count_dst(const long long* __restrict__ ei,
                                                   int* __restrict__ cnt) {
    int e = blockIdx.x * blockDim.x + threadIdx.x;
    if (e < NE) atomicAdd(&cnt[(int)ei[NE + e]], 1);
}

// single-block exclusive scan over NN counts -> noff[0..NN]
__global__ __launch_bounds__(1024) void k_scan(const int* __restrict__ cnt,
                                               int* __restrict__ noff) {
    __shared__ int carry;
    __shared__ int warpsums[32];
    const int lane = threadIdx.x & 31;
    const int wid  = threadIdx.x >> 5;
    if (threadIdx.x == 0) carry = 0;
    __syncthreads();
    for (int base = 0; base < NN; base += 1024) {
        int i = base + threadIdx.x;
        int v = (i < NN) ? cnt[i] : 0;
        int x = v;
#pragma unroll
        for (int o = 1; o < 32; o <<= 1) {
            int y = __shfl_up(x, o, 32);
            if (lane >= o) x += y;
        }
        if (lane == 31) warpsums[wid] = x;
        __syncthreads();
        if (wid == 0) {
            int s = warpsums[lane];
#pragma unroll
            for (int o = 1; o < 32; o <<= 1) {
                int y = __shfl_up(s, o, 32);
                if (lane >= o) s += y;
            }
            warpsums[lane] = s;
        }
        __syncthreads();
        int wpre = (wid > 0) ? warpsums[wid - 1] : 0;
        if (i < NN) noff[i] = carry + wpre + x - v;
        __syncthreads();
        if (threadIdx.x == 0) carry += warpsums[31];
        __syncthreads();
    }
    if (threadIdx.x == 0) noff[NN] = carry;
}

__global__ __launch_bounds__(256) void k_fill(const long long* __restrict__ ei,
                                              const int* __restrict__ noff,
                                              int* __restrict__ cursor,
                                              int* __restrict__ bkt) {
    int e = blockIdx.x * blockDim.x + threadIdx.x;
    if (e >= NE) return;
    int d = (int)ei[NE + e];
    int pos = atomicAdd(&cursor[d], 1);
    bkt[noff[d] + pos] = e;
}

// sort each bucket ascending -> deterministic summation order across replays
__global__ __launch_bounds__(256) void k_sortbkt(const int* __restrict__ noff,
                                                 int* __restrict__ bkt) {
    int n = blockIdx.x * blockDim.x + threadIdx.x;
    if (n >= NN) return;
    int s0 = noff[n], s1 = noff[n + 1];
    for (int i = s0 + 1; i < s1; ++i) {
        int v = bkt[i];
        int j = i - 1;
        while (j >= s0 && bkt[j] > v) { bkt[j + 1] = bkt[j]; --j; }
        bkt[j + 1] = v;
    }
}

// ------------------------------- CSR gather prop ---------------------------
// yout[b,node,:] = alpha * sum_{e: dst==node} nrm[e]*xin[b,src[e],:]
//                  + beta * xbase[b,node,:]
// one wave per (batch, node); lane owns 8 consecutive channels.
__global__ __launch_bounds__(256) void k_gather(const int* __restrict__ bkt,
                                                const int* __restrict__ noff,
                                                const long long* __restrict__ ei,
                                                const float* __restrict__ nrm,
                                                const float* __restrict__ xin,
                                                const float* __restrict__ xbase,
                                                float alpha, float beta,
                                                float* __restrict__ yout) {
    const int wave = threadIdx.x >> 5;
    const int lane = threadIdx.x & 31;
    const int bb   = blockIdx.x / (NN / 8);          // 2500 blocks per batch
    const int node = (blockIdx.x % (NN / 8)) * 8 + wave;
    const int c0   = lane * 8;

    float4 aA = make_float4(0.f, 0.f, 0.f, 0.f);
    float4 aB = make_float4(0.f, 0.f, 0.f, 0.f);

    const int s0 = noff[node], s1 = noff[node + 1];
    for (int j = s0; j < s1; ++j) {
        int e   = bkt[j];
        int src = (int)ei[e];
        float w = nrm[e];
        const float4* xp = (const float4*)(xin + ((long)bb * NN + src) * CD + c0);
        float4 a = xp[0];
        float4 b = xp[1];
        aA.x += w * a.x; aA.y += w * a.y; aA.z += w * a.z; aA.w += w * a.w;
        aB.x += w * b.x; aB.y += w * b.y; aB.z += w * b.z; aB.w += w * b.w;
    }

    float4 r0, r1;
    if (beta != 0.0f) {
        const float4* bp = (const float4*)(xbase + ((long)bb * NN + node) * CD + c0);
        float4 b0 = bp[0], b1 = bp[1];
        r0 = make_float4(alpha * aA.x + beta * b0.x, alpha * aA.y + beta * b0.y,
                         alpha * aA.z + beta * b0.z, alpha * aA.w + beta * b0.w);
        r1 = make_float4(alpha * aB.x + beta * b1.x, alpha * aB.y + beta * b1.y,
                         alpha * aB.z + beta * b1.z, alpha * aB.w + beta * b1.w);
    } else {
        r0 = make_float4(alpha * aA.x, alpha * aA.y, alpha * aA.z, alpha * aA.w);
        r1 = make_float4(alpha * aB.x, alpha * aB.y, alpha * aB.z, alpha * aB.w);
    }
    float4* op = (float4*)(yout + ((long)bb * NN + node) * CD + c0);
    op[0] = r0;
    op[1] = r1;
}

// Wl[k][c_in][c_out] (f32) -> wt[k][c_out][c_in] (bf16), B-fragment friendly
__global__ __launch_bounds__(256) void k_pack_w(const float* __restrict__ Wl,
                                                __bf16* __restrict__ wt) {
    int tid = blockIdx.x * blockDim.x + threadIdx.x;  // KORD*CD*CD = 196608 exact
    int k    = tid / (CD * CD);
    int rem  = tid % (CD * CD);
    int cin  = rem / CD;
    int cout = rem % CD;
    wt[((size_t)k * CD + cout) * CD + cin] =
        (__bf16)Wl[((size_t)k * CD + cin) * CD + cout];
}

// ------------------------------- WMMA GEMM ---------------------------------
// Out[m, n] = sum_k ( X0*W0 + X1*W1 + X2*W2 ) + bias[n]
// block = 64-row M block (4 tiles); wave w owns N tiles w and w+8;
// B fragments held in registers and reused across the 4 M tiles.
__global__ __launch_bounds__(256) void k_cheb_gemm(const float* __restrict__ X0,
                                                   const float* __restrict__ X1,
                                                   const float* __restrict__ X2,
                                                   const __bf16* __restrict__ Wt,
                                                   const float* __restrict__ bias,
                                                   float* __restrict__ out) {
    __shared__ alignas(16) __bf16 ldsA[3][64][32];  // 12 KB

    const int  t    = threadIdx.x;
    const int  wave = t >> 5;
    const int  lane = t & 31;
    const int  half = lane >> 4;
    const int  l15  = lane & 15;
    const long row0 = (long)blockIdx.x * 64;

    // staging map: per matrix, thread t converts 8 consecutive floats
    const int srow = t >> 2;          // 0..63
    const int scol = (t & 3) * 8;     // 0,8,16,24

    v8f acc[4][2];
#pragma unroll
    for (int mt = 0; mt < 4; ++mt) {
        acc[mt][0] = {};
        acc[mt][1] = {};
    }

    const float* Xs[3] = {X0, X1, X2};

    for (int k0 = 0; k0 < CD; k0 += 32) {
#pragma unroll
        for (int m = 0; m < 3; ++m) {
            const float4* gp =
                (const float4*)(Xs[m] + (row0 + srow) * CD + k0 + scol);
            float4 f0 = gp[0];
            float4 f1 = gp[1];
            __bf16* lp = &ldsA[m][srow][scol];
            lp[0] = (__bf16)f0.x; lp[1] = (__bf16)f0.y;
            lp[2] = (__bf16)f0.z; lp[3] = (__bf16)f0.w;
            lp[4] = (__bf16)f1.x; lp[5] = (__bf16)f1.y;
            lp[6] = (__bf16)f1.z; lp[7] = (__bf16)f1.w;
        }
        __syncthreads();

        // B fragments for this k-step: 2 N-tiles x 3 matrices, kept in VGPRs
        v16bf bfrag[2][3];
#pragma unroll
        for (int nt = 0; nt < 2; ++nt) {
            const int n0 = (wave + nt * 8) << 4;
#pragma unroll
            for (int m = 0; m < 3; ++m) {
                const __bf16* bp = Wt + ((size_t)m * CD + n0 + l15) * CD + k0;
                union { v16bf v; v8bf h[2]; } ub;
                ub.h[0] = *(const v8bf*)(bp + half * 8);
                ub.h[1] = *(const v8bf*)(bp + 16 + half * 8);
                bfrag[nt][m] = ub.v;
            }
        }

        // 4 M tiles x 2 N tiles x 3 matrices = 24 WMMAs per wave per k-step
#pragma unroll
        for (int mt = 0; mt < 4; ++mt) {
            const int arow = mt * 16 + l15;
#pragma unroll
            for (int m = 0; m < 3; ++m) {
                union { v16bf v; v8bf h[2]; } ua;
                ua.h[0] = *(const v8bf*)&ldsA[m][arow][half * 8];
                ua.h[1] = *(const v8bf*)&ldsA[m][arow][16 + half * 8];
                acc[mt][0] = __builtin_amdgcn_wmma_f32_16x16x32_bf16(
                    false, ua.v, false, bfrag[0][m], (short)0, acc[mt][0],
                    false, false);
                acc[mt][1] = __builtin_amdgcn_wmma_f32_16x16x32_bf16(
                    false, ua.v, false, bfrag[1][m], (short)0, acc[mt][1],
                    false, false);
            }
        }
        __syncthreads();
    }

    // epilogue: D layout — row = 8*half + r, col = l15 (ISA 7.12.2 C/D 16x16)
#pragma unroll
    for (int nt = 0; nt < 2; ++nt) {
        const int n    = ((wave + nt * 8) << 4) + l15;
        const float bv = bias[n];
#pragma unroll
        for (int mt = 0; mt < 4; ++mt) {
            const v8f a = acc[mt][nt];
#pragma unroll
            for (int r = 0; r < 8; ++r) {
                const long m = row0 + mt * 16 + half * 8 + r;
                out[m * CD + n] = a[r] + bv;
            }
        }
    }
}

// ------------------------------- LayerNorm ---------------------------------
// one wave per row (wave32), lane owns 8 consecutive channels
__global__ __launch_bounds__(256) void k_layernorm(float* __restrict__ out,
                                                   const float* __restrict__ lnw,
                                                   const float* __restrict__ lnb) {
    const int  wave = threadIdx.x >> 5;
    const int  lane = threadIdx.x & 31;
    const long row  = (long)blockIdx.x * 8 + wave;
    float* p = out + row * CD + lane * 8;

    float4 va = *(const float4*)p;
    float4 vb = *(const float4*)(p + 4);
    float s  = va.x + va.y + va.z + va.w + vb.x + vb.y + vb.z + vb.w;
    float sq = va.x * va.x + va.y * va.y + va.z * va.z + va.w * va.w +
               vb.x * vb.x + vb.y * vb.y + vb.z * vb.z + vb.w * vb.w;
#pragma unroll
    for (int o = 16; o >= 1; o >>= 1) {
        s  += __shfl_xor(s, o, 32);
        sq += __shfl_xor(sq, o, 32);
    }
    const float mean = s * (1.0f / CD);
    const float var  = sq * (1.0f / CD) - mean * mean;
    const float inv  = rsqrtf(var + EPSF);

    const int c0 = lane * 8;
    float4 w0 = *(const float4*)(lnw + c0);
    float4 w1 = *(const float4*)(lnw + c0 + 4);
    float4 b0 = *(const float4*)(lnb + c0);
    float4 b1 = *(const float4*)(lnb + c0 + 4);

    va.x = (va.x - mean) * inv * w0.x + b0.x;
    va.y = (va.y - mean) * inv * w0.y + b0.y;
    va.z = (va.z - mean) * inv * w0.z + b0.z;
    va.w = (va.w - mean) * inv * w0.w + b0.w;
    vb.x = (vb.x - mean) * inv * w1.x + b1.x;
    vb.y = (vb.y - mean) * inv * w1.y + b1.y;
    vb.z = (vb.z - mean) * inv * w1.z + b1.z;
    vb.w = (vb.w - mean) * inv * w1.w + b1.w;

    *(float4*)p       = va;
    *(float4*)(p + 4) = vb;
}

// ------------------------------- launcher ----------------------------------

extern "C" void kernel_launch(void* const* d_in, const int* in_sizes, int n_in,
                              void* d_out, int out_size, void* d_ws, size_t ws_size,
                              hipStream_t stream) {
    const float*     feat = (const float*)d_in[0];
    const long long* ei   = (const long long*)d_in[1];
    const float*     W    = (const float*)d_in[2];
    const float*     bvec = (const float*)d_in[3];
    const float*     lnw  = (const float*)d_in[4];
    const float*     lnb  = (const float*)d_in[5];
    float*           out  = (float*)d_out;
    (void)in_sizes; (void)n_in; (void)out_size; (void)ws_size;

    char* ws = (char*)d_ws;
    size_t off = 0;
    auto wsalloc = [&](size_t bytes) -> void* {
        void* p = ws + off;
        off += (bytes + 255) & ~(size_t)255;
        return p;
    };
    float*  deg    = (float*)wsalloc((size_t)NN * 4);
    float*  nrm    = (float*)wsalloc((size_t)NE * 4);
    int*    cnt    = (int*)wsalloc((size_t)NN * 4);
    int*    noff   = (int*)wsalloc((size_t)(NN + 1) * 4);
    int*    cursor = (int*)wsalloc((size_t)NN * 4);
    int*    bkt    = (int*)wsalloc((size_t)NE * 4);
    float*  tx1    = (float*)wsalloc((size_t)NROWS * CD * 4);
    float*  tx2    = (float*)wsalloc((size_t)NROWS * CD * 4);
    __bf16* wt     = (__bf16*)wsalloc((size_t)KORD * CD * CD * 2);

    // only layer L-1 == 1 is live
    const float* Wl   = W    + (size_t)1 * KORD * CD * CD;
    const float* bl   = bvec + CD;
    const float* lnwl = lnw  + CD;
    const float* lnbl = lnb  + CD;

    // 1) degree (by src) + edge normalization
    k_zero_f4<<<(NN / 4 + 255) / 256, 256, 0, stream>>>((float4*)deg, NN / 4);
    k_degree<<<NE / 256, 256, 0, stream>>>(ei, deg);
    k_norm<<<NE / 256, 256, 0, stream>>>(ei, deg, nrm);

    // 2) CSR by dst: count -> scan -> fill -> per-bucket sort (determinism)
    k_zero_i<<<(NN + 255) / 256, 256, 0, stream>>>(cnt, NN);
    k_zero_i<<<(NN + 255) / 256, 256, 0, stream>>>(cursor, NN);
    k_count_dst<<<NE / 256, 256, 0, stream>>>(ei, cnt);
    k_scan<<<1, 1024, 0, stream>>>(cnt, noff);
    k_fill<<<NE / 256, 256, 0, stream>>>(ei, noff, cursor, bkt);
    k_sortbkt<<<(NN + 255) / 256, 256, 0, stream>>>(noff, bkt);

    // 3) Chebyshev basis via gathers (no float atomics):
    //    tx1 = L_hat X ; tx2 = 2 L_hat tx1 - X
    k_gather<<<NROWS / 8, 256, 0, stream>>>(bkt, noff, ei, nrm, feat, feat,
                                            1.0f, 0.0f, tx1);
    k_gather<<<NROWS / 8, 256, 0, stream>>>(bkt, noff, ei, nrm, tx1, feat,
                                            2.0f, -1.0f, tx2);

    // 4) pack transposed bf16 weights for WMMA B fragments
    k_pack_w<<<(KORD * CD * CD) / 256, 256, 0, stream>>>(Wl, wt);

    // 5) fused triple GEMM (bf16 WMMA, f32 accumulate) + bias
    k_cheb_gemm<<<NROWS / 64, 256, 0, stream>>>(feat, tx1, tx2, wt, bl, out);

    // 6) in-place LayerNorm
    k_layernorm<<<NROWS / 8, 256, 0, stream>>>(out, lnwl, lnbl);
}
